// Parking_Trajectory_Planner_29446295781499
// MI455X (gfx1250) — matrix operand order repaired and checked by reading, not compiled, not run on hardware
//
#include <hip/hip_runtime.h>
#include <math.h>

// ---------------- model constants ----------------
#define BATCH      64
#define NSTEP      2
#define NANCH      6
#define SMD        256
#define NMAP       240
#define MAPMAX     256
#define CAR_L      2.5f
#define DLV_C      0.3f
#define DLM0_C     1.0f
#define DLM1_C     0.6f
#define LNEPS      1e-5f
#define ATT_SCALE  0.0625f   // 1/sqrt(256)

// ---------------- output layout (floats) ----------------
#define OFF_PM 0          // (B,2,6,3)   2304
#define OFF_PV 2304       // (B,2,6,3)   2304
#define OFF_VW 4608       // (B,2,6,2,256) 393216
#define OFF_ST 397824     // (B,2,6,2)   1536

// ---------------- workspace layout (floats) ----------------
#define WOFF_A      0               // A[3][2][2][256] : Aq/Ak/Av
#define WOFF_G      3072            // G[2][2][2]
#define WOFF_WVO    3080            // Wvo[2][2][256]
#define WOFF_WIH0T  4104            // [2][512][1024]
#define WOFF_WHH0T  (4104+1048576)  // [2][256][1024]
#define WOFF_WIH1T  (WOFF_WHH0T+524288)
#define WOFF_WHH1T  (WOFF_WIH1T+524288)
#define WOFF_WM1H   (WOFF_WHH1T+524288)   // wm1 as f16: 2*65536 halves = 65536 floats

// ---------------- LDS layout (bytes) ----------------
#define LROW 264                       // padded f16 row stride (halves): 512B data + 16B pad
#define L_TMAT  0                      // 256*264*2 = 135168   t[d][q] f16
#define L_WM1   135168                 // 256*264*2 = 135168   wm1[s][q] f16 (TDM-filled)
#define L_MXY   270336                 // float2[256]  2048
#define L_Z     272384                 // float[256]   1024
#define L_XBUF  273408                 // float[512]   2048
#define L_H0    275456
#define L_C0    276480
#define L_H1    277504
#define L_C1    278528
#define L_RED   279552                 // float[256]
#define L_WVOL  280576                 // float[512]
#define L_WM2L  282624                 // float[256]
#define L_SBUF  283648                 // float[16]
#define SMEM_BYTES 283712

typedef _Float16 v16h __attribute__((ext_vector_type(16)));
typedef _Float16 v8h  __attribute__((ext_vector_type(8)));
typedef float    v8f  __attribute__((ext_vector_type(8)));
typedef unsigned int u32x4 __attribute__((ext_vector_type(4)));
typedef int      i32x4 __attribute__((ext_vector_type(4)));
typedef int      i32x8 __attribute__((ext_vector_type(8)));
union V16U { v16h v; v8h h[2]; };

__device__ __forceinline__ float sigm(float x) { return 1.0f / (1.0f + __expf(-x)); }

__device__ __forceinline__ float block_sum(float v, float* red, int tid) {
  __syncthreads();                 // protect previous use of red
  red[tid] = v; __syncthreads();
  #pragma unroll
  for (int s = 128; s > 0; s >>= 1) {
    if (tid < s) red[tid] += red[tid + s];
    __syncthreads();
  }
  float r = red[0]; __syncthreads();
  return r;
}

__device__ __forceinline__ float block_ln(float v, float* red, int tid) {
  float mu = block_sum(v, red, tid) * (1.0f / 256.0f);
  float d  = v - mu;
  float var = block_sum(d * d, red, tid) * (1.0f / 256.0f);
  return d * rsqrtf(var + LNEPS);
}

// cal_map_last: rotate map into anchor frame, mask >=10, sort by x, mask->0,
// write the (2,256) view for (b,st,k), leave sorted points in Mxy.
__device__ void compute_map_and_view(int tid, int b, int st, int k,
                                     const float* sbuf, float2* Mxy,
                                     const float* mapg, float* outp) {
  __syncthreads();
  float ax = sbuf[0], ay = sbuf[1], th = sbuf[2];
  float c = __cosf(th), s = __sinf(th);
  if (tid < NMAP) {
    float dx = mapg[tid] - ax;
    float dy = mapg[NMAP + tid] - ay;
    float m0 = c * dx + s * dy;
    float m1 = -s * dx + c * dy;
    float dis = sqrtf(m0 * m0 + m1 * m1);
    if (dis >= 10.0f) { m0 = 1e10f; m1 = 1e10f; }
    Mxy[tid] = make_float2(m0, m1);
  } else {
    Mxy[tid] = make_float2(1e30f, 1e30f);   // sentinel sorts to the end
  }
  __syncthreads();
  // bitonic sort 256 elements by x (ascending)
  for (int kk = 2; kk <= 256; kk <<= 1) {
    for (int j = kk >> 1; j > 0; j >>= 1) {
      int ixj = tid ^ j;
      if (ixj > tid) {
        float2 a = Mxy[tid], bb = Mxy[ixj];
        bool up = ((tid & kk) == 0);
        bool sw = up ? (a.x > bb.x) : (a.x < bb.x);
        if (sw) { Mxy[tid] = bb; Mxy[ixj] = a; }
      }
      __syncthreads();
    }
  }
  float2 v = Mxy[tid];
  if (v.x >= 1e9f) v.x = 0.0f;
  if (v.y >= 1e9f) v.y = 0.0f;
  Mxy[tid] = v;
  __syncthreads();
  size_t vb = OFF_VW + ((((size_t)(b * 2 + st)) * 6 + k) * 2) * 256;
  outp[vb + tid]       = v.x;
  outp[vb + 256 + tid] = v.y;
}

// ---------------- setup kernels ----------------
__global__ void setup_proj(const float* wq, const float* wk, const float* wv,
                           const float* wiq, const float* wik, const float* wiv,
                           float* ws) {
  int t = blockIdx.x * blockDim.x + threadIdx.x;
  if (t >= 3072) return;
  int j = t & 255, r = (t >> 8) & 1, st = (t >> 9) & 1, w = t >> 10;
  const float* sml = (w == 0) ? wq : (w == 1) ? wk : wv;    // (2,256,2)
  const float* big = (w == 0) ? wiq : (w == 1) ? wik : wiv; // (2,256,256)
  float acc = 0.0f;
  for (int s = 0; s < SMD; ++s)
    acc += sml[st * 512 + s * 2 + r] * big[st * 65536 + j * 256 + s];
  ws[WOFF_A + ((size_t)((w * 2 + st) * 2 + r)) * 256 + j] = acc;
}

__global__ void setup_gvo(const float* wout, float* ws) {
  int t = blockIdx.x * blockDim.x + threadIdx.x;
  if (t < 1024) {            // Wvo[st][r][j] = sum_d Av[st][r][d]*wout[st][j][d]
    int j = t & 255, r = (t >> 8) & 1, st = t >> 9;
    const float* Av = ws + WOFF_A + ((size_t)((4 + st) * 2 + r)) * 256;
    float acc = 0.0f;
    for (int d = 0; d < SMD; ++d) acc += Av[d] * wout[st * 65536 + j * 256 + d];
    ws[WOFF_WVO + (st * 2 + r) * 256 + j] = acc;
  } else if (t < 1032) {     // G[st][r][c] = sum_d Aq[st][r][d]*Ak[st][c][d]
    int u = t - 1024;
    int c = u & 1, r = (u >> 1) & 1, st = u >> 2;
    const float* Aq = ws + WOFF_A + ((size_t)((0 + st) * 2 + r)) * 256;
    const float* Ak = ws + WOFF_A + ((size_t)((2 + st) * 2 + c)) * 256;
    float acc = 0.0f;
    for (int d = 0; d < SMD; ++d) acc += Aq[d] * Ak[d];
    ws[WOFF_G + st * 4 + r * 2 + c] = acc;
  }
}

__global__ void setup_transpose(const float* wih0, const float* whh0,
                                const float* wih1, const float* whh1,
                                const float* wm1, float* ws) {
  int stride = gridDim.x * blockDim.x;
  int tid0 = blockIdx.x * blockDim.x + threadIdx.x;
  // wih0T: [st][j(512)][r(1024)] <- wih0[st][r][j]
  for (int i = tid0; i < 2 * 512 * 1024; i += stride) {
    int st = i / (512 * 1024), rem = i % (512 * 1024);
    int j = rem >> 10, r = rem & 1023;
    ws[WOFF_WIH0T + i] = wih0[(size_t)st * 524288 + (size_t)r * 512 + j];
  }
  for (int i = tid0; i < 2 * 256 * 1024; i += stride) {
    int st = i / (256 * 1024), rem = i % (256 * 1024);
    int j = rem >> 10, r = rem & 1023;
    ws[WOFF_WHH0T + i] = whh0[(size_t)st * 262144 + (size_t)r * 256 + j];
    ws[WOFF_WIH1T + i] = wih1[(size_t)st * 262144 + (size_t)r * 256 + j];
    ws[WOFF_WHH1T + i] = whh1[(size_t)st * 262144 + (size_t)r * 256 + j];
  }
  // wm1 -> f16 (raw bytes for the TDM tile DMA)
  _Float16* wm1h = (_Float16*)(ws + WOFF_WM1H);
  for (int i = tid0; i < 2 * 65536; i += stride)
    wm1h[i] = (_Float16)wm1[i];
}

// ---------------- main fused kernel: 1 workgroup == 1 batch element ----------------
__global__ __launch_bounds__(256)
void planner_main(const float* inp, const float* mapg, const float* endg,
                  const float* wencg, const float* wm2g,
                  const float* wmeang, const float* wvarg,
                  const float* ws, float* outp) {
  extern __shared__ __align__(16) char smem[];
  _Float16* tmat = (_Float16*)(smem + L_TMAT);
  _Float16* wm1l = (_Float16*)(smem + L_WM1);
  float2*   Mxy  = (float2*)  (smem + L_MXY);
  float*    z    = (float*)   (smem + L_Z);
  float*    xbuf = (float*)   (smem + L_XBUF);
  float*    h0   = (float*)   (smem + L_H0);
  float*    c0   = (float*)   (smem + L_C0);
  float*    h1   = (float*)   (smem + L_H1);
  float*    c1   = (float*)   (smem + L_C1);
  float*    red  = (float*)   (smem + L_RED);
  float*    wvol = (float*)   (smem + L_WVOL);
  float*    wm2l = (float*)   (smem + L_WM2L);
  float*    sbuf = (float*)   (smem + L_SBUF);

  const int b = blockIdx.x;
  const int tid = threadIdx.x;
  const int lane = tid & 31;
  const int wave = tid >> 5;
  const _Float16* wm1h = (const _Float16*)(ws + WOFF_WM1H);

  if (tid < 3) {
    sbuf[tid]     = inp[b * 3 + tid];   // anchor_last
    sbuf[3 + tid] = 0.01f;              // var_last
  }
  __syncthreads();

  float e0 = endg[0], e1 = endg[1], e2 = endg[2];

  for (int st = 0; st < NSTEP; ++st) {
    const float dir = (st == 0) ? 1.0f : -1.0f;

    // ---- per-step LDS preload ----
    // wm1 (f16) via Tensor Data Mover: 256x256 2-byte tile, LDS padding
    // 4 DWORDs every 128 DWORDs -> 528B row stride == LROW halves.
    // TDM ignores EXEC, so gate with a *scalar* uniform branch (wave 0 only).
    if (__builtin_amdgcn_readfirstlane(tid) == 0) {
      unsigned int ldsoff = (unsigned int)(unsigned long long)(void*)wm1l;
      unsigned long long gaddr =
          (unsigned long long)(const void*)(wm1h + (size_t)st * 65536);
      u32x4 g0;
      g0[0] = 1u;                                   // count=1, user mode
      g0[1] = ldsoff;                               // lds_addr
      g0[2] = (unsigned int)gaddr;                  // global_addr[31:0]
      g0[3] = (unsigned int)(gaddr >> 32) | (2u << 30); // addr[56:32] | type=2
      i32x8 g1;
      g1[0] = (int)((1u << 16)   // data_size = 2 bytes
                  | (1u << 20)   // pad_enable
                  | (6u << 22)   // pad_interval: every 128 DWORDs (512B)
                  | (3u << 25)); // pad_amount: 4 DWORDs (16B)
      g1[1] = (int)(256u << 16); // tensor_dim0 = 256 (low 16 in [63:48])
      g1[2] = (int)(256u << 16); // tensor_dim1 = 256 (low 16 in [111:96... bits 95:80])
      g1[3] = (int)(256u << 16); // tile_dim0 = 256 ([127:112])
      g1[4] = 256;               // tile_dim1 = 256 ([143:128]); tile_dim2 = 0
      g1[5] = 256;               // tensor_dim0_stride = 256 (data_size units)
      g1[6] = 0;
      g1[7] = 0;
      i32x4 g2; g2[0] = 0; g2[1] = 0; g2[2] = 0; g2[3] = 0;
      i32x4 g3 = g2;
#if defined(__clang_major__) && (__clang_major__ >= 23)
      i32x8 g4;
      g4[0]=0; g4[1]=0; g4[2]=0; g4[3]=0; g4[4]=0; g4[5]=0; g4[6]=0; g4[7]=0;
      __builtin_amdgcn_tensor_load_to_lds(g0, g1, g2, g3, g4, 0);
#else
      __builtin_amdgcn_tensor_load_to_lds(g0, g1, g2, g3, 0);
#endif
      __builtin_amdgcn_s_wait_tensorcnt(0);
    }
    wvol[tid]       = ws[WOFF_WVO + (st * 2 + 0) * 256 + tid];
    wvol[256 + tid] = ws[WOFF_WVO + (st * 2 + 1) * 256 + tid];
    wm2l[tid]       = wm2g[st * 256 + tid];
    h0[tid] = 1.0f; c0[tid] = 1.0f; h1[tid] = 1.0f; c1[tid] = 1.0f;
    float g00 = ws[WOFF_G + st * 4 + 0], g01 = ws[WOFF_G + st * 4 + 1];
    float g10 = ws[WOFF_G + st * 4 + 2], g11 = ws[WOFF_G + st * 4 + 3];
    // k=0 outputs
    if (tid < 3) {
      size_t base = ((size_t)(b * 2 + st) * 6 + 0) * 3;
      outp[OFF_PM + base + tid] = sbuf[tid];
      outp[OFF_PV + base + tid] = sbuf[3 + tid];
    }
    if (tid < 2)
      outp[OFF_ST + ((size_t)(b * 2 + st) * 6 + 5) * 2 + tid] = 0.0f;  // states k=5
    __syncthreads();
    compute_map_and_view(tid, b, st, 0, sbuf, Mxy, mapg, outp);

    for (int t = 0; t < NANCH - 1; ++t) {
      // ---- enc_anchor: feat(9) @ w_enc^T, then LN ----
      float a0 = sbuf[0], a1 = sbuf[1], a2 = sbuf[2];
      float v0 = sbuf[3], v1 = sbuf[4], v2 = sbuf[5];
      float feat[9] = {a0, a1, a2, a0 - e0, a1 - e1, a2 - e2, v0, v1, v2};
      const float* wrow = wencg + (size_t)st * 2304 + (size_t)tid * 9;
      float ea = 0.0f;
      #pragma unroll
      for (int kx = 0; kx < 9; ++kx) ea += feat[kx] * wrow[kx];
      xbuf[tid] = block_ln(ea, red, tid);

      // ---- rank-2 attention row (thread tid = query row), online softmax ----
      float mix = Mxy[tid].x, miy = Mxy[tid].y;
      float ci0 = g00 * mix + g10 * miy;
      float ci1 = g01 * mix + g11 * miy;
      float mrun = -3.0e38f, lrun = 0.0f, p0 = 0.0f, p1 = 0.0f;
      for (int j = 0; j < MAPMAX; ++j) {
        float2 mj = Mxy[j];
        float sc = (ci0 * mj.x + ci1 * mj.y) * ATT_SCALE;
        if (sc > mrun) {
          float cr = __expf(mrun - sc);
          lrun *= cr; p0 *= cr; p1 *= cr; mrun = sc;
        }
        float ex = __expf(sc - mrun);
        lrun += ex; p0 += ex * mj.x; p1 += ex * mj.y;
      }
      float inv = 1.0f / lrun; p0 *= inv; p1 *= inv;
      // o[tid][d] = p0*Wvo[0][d] + p1*Wvo[1][d];  t[d][tid] = relu(o)
      for (int d = 0; d < SMD; ++d) {
        float ov = p0 * wvol[d] + p1 * wvol[256 + d];
        tmat[d * LROW + tid] = (_Float16)fmaxf(ov, 0.0f);
      }
      z[tid] = 0.0f;
      __syncthreads();

      // ---- WMMA GEMM: u = t @ wm1^T, fused relu(u)*wm2 -> z ----
      const int sub = (lane >> 4) & 1;   // 0: lanes 0-15, 1: lanes 16-31
      for (int dbi = 0; dbi < 2; ++dbi) {
        int db = wave * 2 + dbi;
        float zacc[8];
        #pragma unroll
        for (int r = 0; r < 8; ++r) zacc[r] = 0.0f;
        const _Float16* arow = tmat + (size_t)(db * 16 + (lane & 15)) * LROW;
        for (int sb = 0; sb < 16; ++sb) {
          const _Float16* brow = wm1l + (size_t)(sb * 16 + (lane & 15)) * LROW;
          v8f C = {0.f, 0.f, 0.f, 0.f, 0.f, 0.f, 0.f, 0.f};
          #pragma unroll
          for (int kb = 0; kb < 8; ++kb) {
            V16U A, Bm;
            const v8h* pa = (const v8h*)(arow + kb * 32 + sub * 8);
            A.h[0] = pa[0];        // K 0..7   (or 8..15)
            A.h[1] = pa[2];        // K 16..23 (or 24..31)
            const v8h* pb = (const v8h*)(brow + kb * 32 + sub * 8);
            Bm.h[0] = pb[0];       // K 0..7   (or 8..15)
            Bm.h[1] = pb[2];       // K 16..23 (or 24..31)
            C = __builtin_amdgcn_wmma_f32_16x16x32_f16(
                    false, A.v, false, Bm.v, (short)0, C, false, false);
          }
          float w2 = wm2l[sb * 16 + (lane & 15)];
          #pragma unroll
          for (int r = 0; r < 8; ++r) zacc[r] += fmaxf(C[r], 0.0f) * w2;
        }
        int rbase = db * 16 + sub * 8;
        #pragma unroll
        for (int r = 0; r < 8; ++r) atomicAdd(&z[rbase + r], zacc[r]);
      }
      __syncthreads();

      // ---- enc_map = LN(z) ----
      float em = block_ln(z[tid], red, tid);
      xbuf[256 + tid] = em;
      __syncthreads();

      // ---- LSTM layer 0 (coalesced via transposed weights, L2 prefetch) ----
      {
        const float* W = ws + WOFF_WIH0T + (size_t)st * 524288;
        const float* H = ws + WOFF_WHH0T + (size_t)st * 262144;
        float gi = 0.f, gf = 0.f, gg = 0.f, go = 0.f;
        for (int j = 0; j < 512; ++j) {
          float xv = xbuf[j];
          const float* p = W + (size_t)j * 1024 + tid;
          if ((j & 63) == 0) __builtin_prefetch(p + 65536, 0, 0);
          gi += xv * p[0]; gf += xv * p[256]; gg += xv * p[512]; go += xv * p[768];
        }
        for (int j = 0; j < 256; ++j) {
          float hv = h0[j];
          const float* p = H + (size_t)j * 1024 + tid;
          if ((j & 63) == 0) __builtin_prefetch(p + 65536, 0, 0);
          gi += hv * p[0]; gf += hv * p[256]; gg += hv * p[512]; go += hv * p[768];
        }
        float cn = sigm(gf) * c0[tid] + sigm(gi) * tanhf(gg);
        float hn = sigm(go) * tanhf(cn);
        __syncthreads();
        h0[tid] = hn; c0[tid] = cn;
        __syncthreads();
      }
      // ---- LSTM layer 1 ----
      {
        const float* W = ws + WOFF_WIH1T + (size_t)st * 262144;
        const float* H = ws + WOFF_WHH1T + (size_t)st * 262144;
        float gi = 0.f, gf = 0.f, gg = 0.f, go = 0.f;
        for (int j = 0; j < 256; ++j) {
          float xv = h0[j];
          const float* p = W + (size_t)j * 1024 + tid;
          if ((j & 63) == 0) __builtin_prefetch(p + 65536, 0, 0);
          gi += xv * p[0]; gf += xv * p[256]; gg += xv * p[512]; go += xv * p[768];
        }
        for (int j = 0; j < 256; ++j) {
          float hv = h1[j];
          const float* p = H + (size_t)j * 1024 + tid;
          if ((j & 63) == 0) __builtin_prefetch(p + 65536, 0, 0);
          gi += hv * p[0]; gf += hv * p[256]; gg += hv * p[512]; go += hv * p[768];
        }
        float cn = sigm(gf) * c1[tid] + sigm(gi) * tanhf(gg);
        float hn = sigm(go) * tanhf(cn);
        __syncthreads();
        h1[tid] = hn; c1[tid] = cn;
        __syncthreads();
      }

      // ---- head: mn = LN(h1), raw = relu(mn)@wmean^T, vr = sigmoid(relu(mn)@wvar^T) ----
      float mn = block_ln(h1[tid], red, tid);
      float rm = fmaxf(mn, 0.0f);
      float raw0 = block_sum(rm * wmeang[st * 512 + tid],        red, tid);
      float raw1 = block_sum(rm * wmeang[st * 512 + 256 + tid],  red, tid);
      float dv0  = block_sum(rm * wvarg[st * 768 + tid],         red, tid);
      float dv1  = block_sum(rm * wvarg[st * 768 + 256 + tid],   red, tid);
      float dv2  = block_sum(rm * wvarg[st * 768 + 512 + tid],   red, tid);

      if (tid == 0) {
        float s0 = sigm(raw0) * DLM0_C;
        float s1 = tanhf(raw1) * DLM1_C;
        // bicycle-model anchor update
        float sdist = dir * s0;
        float tp = __tanf(s1);
        float dth = sdist * tp / CAR_L;
        float kk = CAR_L / tp;
        float th = sbuf[2];
        float nx = sbuf[0] + kk * (__sinf(th + dth) - __sinf(th));
        float ny = sbuf[1] - kk * (__cosf(th + dth) - __cosf(th));
        float nth = th + dth;
        float vr[3] = {sigm(dv0), sigm(dv1), sigm(dv2)};
        size_t kb = ((size_t)(b * 2 + st) * 6 + (t + 1));
        outp[OFF_PM + kb * 3 + 0] = nx;
        outp[OFF_PM + kb * 3 + 1] = ny;
        outp[OFF_PM + kb * 3 + 2] = nth;
        #pragma unroll
        for (int q = 0; q < 3; ++q) {
          float sg = (vr[q] > 0.f) ? 1.f : ((vr[q] < 0.f) ? -1.f : 0.f);
          float vn = vr[q] * DLV_C + 0.1f * sg;
          outp[OFF_PV + kb * 3 + q] = vn;
          sbuf[3 + q] = vn;
        }
        size_t sbk = ((size_t)(b * 2 + st) * 6 + t) * 2;
        outp[OFF_ST + sbk + 0] = s0;
        outp[OFF_ST + sbk + 1] = s1;
        sbuf[0] = nx; sbuf[1] = ny; sbuf[2] = nth;
      }
      __syncthreads();
      compute_map_and_view(tid, b, st, t + 1, sbuf, Mxy, mapg, outp);
    }
  }
}

// ---------------- launcher ----------------
extern "C" void kernel_launch(void* const* d_in, const int* in_sizes, int n_in,
                              void* d_out, int out_size, void* d_ws, size_t ws_size,
                              hipStream_t stream) {
  const float* inp   = (const float*)d_in[0];
  const float* mapg  = (const float*)d_in[1];
  const float* endg  = (const float*)d_in[2];
  const float* wenc  = (const float*)d_in[3];
  const float* wq    = (const float*)d_in[4];
  const float* wk    = (const float*)d_in[5];
  const float* wv    = (const float*)d_in[6];
  const float* wiq   = (const float*)d_in[7];
  const float* wik   = (const float*)d_in[8];
  const float* wiv   = (const float*)d_in[9];
  const float* wout  = (const float*)d_in[10];
  const float* wm1   = (const float*)d_in[11];
  const float* wm2   = (const float*)d_in[12];
  const float* wih0  = (const float*)d_in[13];
  const float* whh0  = (const float*)d_in[14];
  const float* wih1  = (const float*)d_in[15];
  const float* whh1  = (const float*)d_in[16];
  const float* wmean = (const float*)d_in[17];
  const float* wvar  = (const float*)d_in[18];
  float* wsf  = (float*)d_ws;
  float* outp = (float*)d_out;

  setup_proj<<<12, 256, 0, stream>>>(wq, wk, wv, wiq, wik, wiv, wsf);
  setup_gvo<<<5, 256, 0, stream>>>(wout, wsf);
  setup_transpose<<<1024, 256, 0, stream>>>(wih0, whh0, wih1, whh1, wm1, wsf);

  (void)hipFuncSetAttribute((const void*)planner_main,
                            hipFuncAttributeMaxDynamicSharedMemorySize, SMEM_BYTES);
  planner_main<<<BATCH, 256, SMEM_BYTES, stream>>>(
      inp, mapg, endg, wenc, wm2, wmean, wvar, wsf, outp);
}
